// TGATLayer_22239340659361
// MI455X (gfx1250) — matrix operand order repaired and checked by reading, not compiled, not run on hardware
//
#include <hip/hip_runtime.h>
#include <math.h>

#define NNODES   10000
#define NEDGES   320000
#define IN_DIMK  512
#define FEAT_DIM 128
#define OUT_DIM  64
#define HEADS    8
#define NHID     512          // OUT_DIM*HEADS
#define NEG_SLOPE 0.01f

typedef _Float16 v16h __attribute__((ext_vector_type(16)));
typedef _Float16 v8h  __attribute__((ext_vector_type(8)));
typedef float    v8f  __attribute__((ext_vector_type(8)));
typedef float    v4f  __attribute__((ext_vector_type(4)));

// ---------------------------------------------------------------- utilities
__global__ void k_zero_counts(int* __restrict__ counts) {
    int t = blockIdx.x * blockDim.x + threadIdx.x;
    if (t < NNODES) counts[t] = 0;
}

// V[k,h] = sum_d w_feat[k, h*64+d] * a_feat[d]   (fold the 42-GFLOP GEMM away)
__global__ void k_fold_v(const float* __restrict__ w_feat,
                         const float* __restrict__ w_attn,
                         float* __restrict__ V) {
    int t = blockIdx.x * blockDim.x + threadIdx.x;
    if (t >= FEAT_DIM * HEADS) return;
    int k = t >> 3, h = t & 7;
    const float* wrow = w_feat + k * NHID + h * OUT_DIM;
    const float* af   = w_attn + 2 * OUT_DIM;
    float acc = 0.f;
    #pragma unroll 8
    for (int d = 0; d < OUT_DIM; ++d) acc += wrow[d] * af[d];
    V[t] = acc;
}

// one-off: WfcT[n][k] = (fp16) w_fc[k][n] — LDS tile transpose, coalesced both ways.
// Moves the B transpose out of the GEMM hot loop (kills the ds_store_b16 scatter).
__global__ __launch_bounds__(256) void k_conv_wfc(const float* __restrict__ Wfc,
                                                  _Float16* __restrict__ WfcT) {
    __shared__ float tile[32][33];
    int bx = blockIdx.x * 32;   // n base
    int by = blockIdx.y * 32;   // k base
    int tx = threadIdx.x & 31, ty = threadIdx.x >> 5;   // 32 x 8
    #pragma unroll
    for (int j = 0; j < 32; j += 8)
        tile[ty + j][tx] = Wfc[(size_t)(by + ty + j) * NHID + bx + tx];
    __syncthreads();
    #pragma unroll
    for (int j = 0; j < 32; j += 8)
        WfcT[(size_t)(bx + ty + j) * IN_DIMK + by + tx] = (_Float16)tile[tx][ty + j];
}

// ---------------------------------------------------------------- z = h @ w_fc
// fp16 inputs / fp32 accumulate via v_wmma_f32_16x16x32_f16, software-pipelined
// global->register->LDS staging; B comes pre-transposed+converted (WfcT).
#define BM 128
#define BN 128
#define KT 32
#define LDA 40   // padded half strides: 40*2B = 80B -> conflict-free b128 frag loads
#define LDB 40

__global__ __launch_bounds__(256) void k_gemm_z(const float* __restrict__ H,
                                                const _Float16* __restrict__ WfcT,
                                                float* __restrict__ Z) {
    __shared__ __align__(16) _Float16 sA[BM * LDA];
    __shared__ __align__(16) _Float16 sB[BN * LDB];

    const int tid   = threadIdx.x;
    const int lane  = tid & 31;
    const int wave  = tid >> 5;
    const int wm    = (wave & 3) * 32;   // wave row offset in block tile
    const int wn    = (wave >> 2) * 64;  // wave col offset in block tile
    const int m0    = blockIdx.y * BM;
    const int n0    = blockIdx.x * BN;
    const int lrow  = lane & 15;
    const int lhalf = lane >> 4;         // 0: lanes 0-15, 1: lanes 16-31

    // --- loop-invariant staging addresses --------------------------------
    const float*    aptr[4];  _Float16* ast[4];
    const _Float16* bptr[2];  _Float16* bst[2];
    #pragma unroll
    for (int it = 0; it < 4; ++it) {
        int i = tid + it * 256;          // A float4 slot
        int r = i >> 3;
        int q = (i & 7) * 4;
        int gr = m0 + r; if (gr > NNODES - 1) gr = NNODES - 1;
        aptr[it] = H + (size_t)gr * IN_DIMK + q;          // + k0 per tile
        ast[it]  = sA + r * LDA + q;
    }
    #pragma unroll
    for (int it = 0; it < 2; ++it) {
        int s2 = tid * 2 + it;           // B v8h slot (512 total)
        int n  = s2 >> 2;                // 0..127
        int kh = (s2 & 3) * 8;           // 0,8,16,24
        bptr[it] = WfcT + (size_t)(n0 + n) * IN_DIMK + kh;  // + k0 per tile
        bst[it]  = sB + n * LDB + kh;
    }

    v8f acc[2][4];
    #pragma unroll
    for (int i = 0; i < 2; ++i)
        #pragma unroll
        for (int j = 0; j < 4; ++j) acc[i][j] = (v8f){0,0,0,0,0,0,0,0};

    // --- prologue: stage tile 0 into registers ---------------------------
    v4f ta[4];
    v8h tb[2];
    #pragma unroll
    for (int it = 0; it < 4; ++it) ta[it] = *(const v4f*)(aptr[it]);
    #pragma unroll
    for (int it = 0; it < 2; ++it) tb[it] = *(const v8h*)(bptr[it]);

    for (int k0 = 0; k0 < IN_DIMK; k0 += KT) {
        // convert/store the staged registers to LDS
        #pragma unroll
        for (int it = 0; it < 4; ++it) {
            _Float16* d = ast[it];
            d[0] = (_Float16)ta[it].x;
            d[1] = (_Float16)ta[it].y;
            d[2] = (_Float16)ta[it].z;
            d[3] = (_Float16)ta[it].w;
        }
        #pragma unroll
        for (int it = 0; it < 2; ++it)
            *(v8h*)bst[it] = tb[it];
        // issue next tile's global loads; they complete during compute below
        if (k0 + KT < IN_DIMK) {
            #pragma unroll
            for (int it = 0; it < 4; ++it)
                ta[it] = *(const v4f*)(aptr[it] + (k0 + KT));
            #pragma unroll
            for (int it = 0; it < 2; ++it)
                tb[it] = *(const v8h*)(bptr[it] + (k0 + KT));
        }
        __syncthreads();

        // fragments per the 16-bit A 16x32 / B 32x16 VGPR layouts
        v16h afr[2], bfr[4];
        #pragma unroll
        for (int ms = 0; ms < 2; ++ms) {
            int row = wm + ms * 16 + lrow;
            v8h lo = *(const v8h*)(sA + row * LDA + lhalf * 8);       // K c..c+7
            v8h hi = *(const v8h*)(sA + row * LDA + lhalf * 8 + 16);  // K c+16..c+23
            #pragma unroll
            for (int j = 0; j < 8; ++j) { afr[ms][j] = lo[j]; afr[ms][8+j] = hi[j]; }
        }
        #pragma unroll
        for (int ns = 0; ns < 4; ++ns) {
            int rowb = wn + ns * 16 + lrow;                            // column N
            v8h lo = *(const v8h*)(sB + rowb * LDB + lhalf * 8);       // K c..c+7
            v8h hi = *(const v8h*)(sB + rowb * LDB + lhalf * 8 + 16);  // K c+16..c+23
            #pragma unroll
            for (int j = 0; j < 8; ++j) { bfr[ns][j] = lo[j]; bfr[ns][8+j] = hi[j]; }
        }
        #pragma unroll
        for (int ms = 0; ms < 2; ++ms)
            #pragma unroll
            for (int ns = 0; ns < 4; ++ns)
                acc[ms][ns] = __builtin_amdgcn_wmma_f32_16x16x32_f16(
                    false, afr[ms], false, bfr[ns], (short)0, acc[ms][ns], false, false);
        __syncthreads();
    }

    // C/D layout: element r -> M = r + lhalf*8, N = lane%16
    #pragma unroll
    for (int ms = 0; ms < 2; ++ms)
        #pragma unroll
        for (int ns = 0; ns < 4; ++ns) {
            int col = n0 + wn + ns * 16 + lrow;
            #pragma unroll
            for (int r = 0; r < 8; ++r) {
                int gm = m0 + wm + ms * 16 + r + lhalf * 8;
                if (gm < NNODES) Z[(size_t)gm * NHID + col] = acc[ms][ns][r];
            }
        }
}

// ---------------------------------------------------------------- node scores from z
__global__ void k_node_scores(const float* __restrict__ Z,
                              const float* __restrict__ w_attn,
                              float* __restrict__ s_src,
                              float* __restrict__ s_dst) {
    int t = blockIdx.x * blockDim.x + threadIdx.x;
    if (t >= NNODES * HEADS) return;
    int n = t >> 3, h = t & 7;
    const float* zr = Z + (size_t)n * NHID + h * OUT_DIM;
    float as = 0.f, ad = 0.f;
    #pragma unroll 8
    for (int d = 0; d < OUT_DIM; ++d) {
        float v = zr[d];
        as += v * w_attn[d];
        ad += v * w_attn[OUT_DIM + d];
    }
    s_src[t] = as;
    s_dst[t] = ad;
}

// ---------------------------------------------------------------- edge scores + degree hist
#define BE 64
__global__ __launch_bounds__(256) void k_edge_scores(
    const float* __restrict__ dep, const int* __restrict__ src,
    const int* __restrict__ dst, const float* __restrict__ V,
    const float* __restrict__ s_src, const float* __restrict__ s_dst,
    float* __restrict__ escore, int* __restrict__ counts) {
    __shared__ float sdep[BE * 129];           // +1 pad -> conflict-free
    __shared__ float sV[FEAT_DIM * HEADS];
    int tid = threadIdx.x;
    int e0  = blockIdx.x * BE;                 // NEDGES % BE == 0
    for (int i = tid; i < FEAT_DIM * HEADS; i += 256) sV[i] = V[i];
    // phase 1: 8 loads in flight (this kernel streams the 164 MB dep_emb)
    v4f t[8];
    #pragma unroll
    for (int it = 0; it < 8; ++it) {
        int f = (tid + it * 256) * 4;
        t[it] = *(const v4f*)(dep + (size_t)e0 * FEAT_DIM + f);
    }
    // phase 2: spill to LDS
    #pragma unroll
    for (int it = 0; it < 8; ++it) {
        int f = (tid + it * 256) * 4;
        int el = f >> 7, kk = f & 127;
        sdep[el*129 + kk + 0] = t[it].x;
        sdep[el*129 + kk + 1] = t[it].y;
        sdep[el*129 + kk + 2] = t[it].z;
        sdep[el*129 + kk + 3] = t[it].w;
    }
    __syncthreads();
    #pragma unroll
    for (int pp = 0; pp < 2; ++pp) {
        int p  = tid + pp * 256;
        int el = p >> 3, h = p & 7;
        const float* dr = sdep + el * 129;
        float acc = 0.f;
        #pragma unroll 8
        for (int kk = 0; kk < FEAT_DIM; ++kk) acc += dr[kk] * sV[kk * HEADS + h];
        int e = e0 + el;
        int sidx = src[e], didx = dst[e];
        float v = acc + s_src[sidx * HEADS + h] + s_dst[didx * HEADS + h];
        v = (v >= 0.f) ? v : NEG_SLOPE * v;    // leaky relu
        escore[(size_t)e * HEADS + h] = v;
        if (h == 0) atomicAdd(&counts[didx], 1);
    }
}

// ---------------------------------------------------------------- exclusive scan -> CSR starts
__global__ __launch_bounds__(1024) void k_scan(const int* __restrict__ counts,
                                               int* __restrict__ starts) {
    __shared__ int part[1024];
    int t = threadIdx.x;
    const int CHK = (NNODES + 1023) / 1024;    // 10
    int base = t * CHK;
    int sum = 0;
    for (int i = 0; i < CHK; ++i) { int idx = base + i; if (idx < NNODES) sum += counts[idx]; }
    part[t] = sum;
    __syncthreads();
    for (int off = 1; off < 1024; off <<= 1) { // Hillis-Steele inclusive scan
        int v = (t >= off) ? part[t - off] : 0;
        __syncthreads();
        part[t] += v;
        __syncthreads();
    }
    int excl = part[t] - sum;
    for (int i = 0; i < CHK; ++i) {
        int idx = base + i;
        if (idx < NNODES) { starts[idx] = excl; excl += counts[idx]; }
    }
    if (t == 0) starts[NNODES] = NEDGES;
}

// ---------------------------------------------------------------- segment softmax + aggregate
#define CH 64
__global__ __launch_bounds__(512) void k_aggregate(
    const float* __restrict__ Z, const float* __restrict__ escore,
    const int* __restrict__ starts, const int* __restrict__ src,
    float* __restrict__ out) {
    __shared__ unsigned smaxu[HEADS];
    __shared__ float smax[HEADS], sden[HEADS];
    __shared__ float salpha[CH * HEADS];
    __shared__ int ssrc[CH];
    int n   = blockIdx.x;
    int tid = threadIdx.x;
    int s = starts[n], e_end = starts[n + 1];
    int deg = e_end - s;
    if (tid < HEADS) { smaxu[tid] = 0x007FFFFFu; sden[tid] = 0.f; }  // enc(-inf)
    __syncthreads();
    // per-head max via order-preserving uint encoding
    for (int i = tid; i < deg * HEADS; i += 512) {
        float v = escore[(size_t)(s + (i >> 3)) * HEADS + (i & 7)];
        unsigned u = __float_as_uint(v);
        u = (u & 0x80000000u) ? ~u : (u | 0x80000000u);
        atomicMax(&smaxu[i & 7], u);
    }
    __syncthreads();
    if (tid < HEADS) {
        unsigned u = smaxu[tid];
        u = (u & 0x80000000u) ? (u & 0x7FFFFFFFu) : ~u;
        smax[tid] = (deg > 0) ? __uint_as_float(u) : 0.f;  // empty segment -> 0
    }
    __syncthreads();
    for (int i = tid; i < deg * HEADS; i += 512) {
        int h = i & 7;
        float v = escore[(size_t)(s + (i >> 3)) * HEADS + h];
        atomicAdd(&sden[h], expf(v - smax[h]));
    }
    __syncthreads();
    const int h = tid >> 6;                    // thread -> (head, dim) output element
    float accv = 0.f;
    for (int c0 = 0; c0 < deg; c0 += CH) {
        int cn = min(CH, deg - c0);
        for (int i = tid; i < cn * HEADS; i += 512) {
            int hh = i & 7;
            float v = escore[(size_t)(s + c0 + (i >> 3)) * HEADS + hh];
            salpha[i] = expf(v - smax[hh]) / fmaxf(sden[hh], 1e-30f);
        }
        for (int i = tid; i < cn; i += 512) ssrc[i] = src[s + c0 + i];
        __syncthreads();
        for (int c = 0; c < cn; ++c)           // coalesced z-row reads (L2-resident)
            accv += salpha[c * HEADS + h] * Z[(size_t)ssrc[c] * NHID + tid];
        __syncthreads();
    }
    out[(size_t)n * NHID + tid] = accv;
}

// ---------------------------------------------------------------- launch
extern "C" void kernel_launch(void* const* d_in, const int* in_sizes, int n_in,
                              void* d_out, int out_size, void* d_ws, size_t ws_size,
                              hipStream_t stream) {
    (void)in_sizes; (void)n_in; (void)out_size; (void)ws_size;
    const float* H      = (const float*)d_in[0];
    const float* dep    = (const float*)d_in[1];
    const int*   src    = (const int*)  d_in[2];
    const int*   dst    = (const int*)  d_in[3];
    const float* w_fc   = (const float*)d_in[4];
    const float* w_feat = (const float*)d_in[5];
    const float* w_attn = (const float*)d_in[6];
    float* out = (float*)d_out;

    char* ws = (char*)d_ws;                     // workspace layout (bytes)
    float*     Z      = (float*)(ws);               // 10000*512*4  = 20,480,000
    float*     escore = (float*)(ws + 20480000);    // 320000*8*4   = 10,240,000
    float*     s_src  = (float*)(ws + 30720000);    // 10000*8*4
    float*     s_dst  = (float*)(ws + 31040000);    // 10000*8*4
    float*     V      = (float*)(ws + 31360000);    // 128*8*4
    int*       counts = (int*)  (ws + 31364096);    // 10000*4
    int*       starts = (int*)  (ws + 31404096);    // 10001*4
    _Float16*  WfcT   = (_Float16*)(ws + 31444112); // 512*512*2 (total ~32 MB)

    k_zero_counts<<<(NNODES + 255) / 256, 256, 0, stream>>>(counts);
    k_fold_v<<<(FEAT_DIM * HEADS + 255) / 256, 256, 0, stream>>>(w_feat, w_attn, V);
    k_conv_wfc<<<dim3(NHID / 32, IN_DIMK / 32), 256, 0, stream>>>(w_fc, WfcT);
    k_gemm_z<<<dim3(NHID / BN, (NNODES + BM - 1) / BM), 256, 0, stream>>>(H, WfcT, Z);
    k_node_scores<<<(NNODES * HEADS + 255) / 256, 256, 0, stream>>>(Z, w_attn, s_src, s_dst);
    k_edge_scores<<<NEDGES / BE, 256, 0, stream>>>(dep, src, dst, V, s_src, s_dst, escore, counts);
    k_scan<<<1, 1024, 0, stream>>>(counts, starts);
    k_aggregate<<<NNODES, 512, 0, stream>>>(Z, escore, starts, src, out);
}
